// FFResidualFCTPProjection_51264729645466
// MI455X (gfx1250) — matrix operand order repaired and checked by reading, not compiled, not run on hardware
//
#include <hip/hip_runtime.h>
#include <math.h>

typedef __bf16 bf16;
typedef __attribute__((ext_vector_type(16))) __bf16 v16bf;
typedef __attribute__((ext_vector_type(8)))  float  v8f;

// LDS strides (elements)
#define XSTR 488   // bf16, 16 rows of 480 (+8 pad)
#define MSTR 968   // bf16, 16 rows of 960 (+8 pad)  [scal 0..255 | v1 256..639 | v2 640..959]
#define GSTR 200   // bf16, 16 rows of 192 gates (+8 pad)
#define SSTR 16    // f32 per node: [0..7] partials, [8]=mu [9]=rsq0 [10]=rsq1 [11]=rsq2 [12]=attr

// per-wave LDS bytes: xs 15616 + mid 30976 + gates 6400 + stats 1024
#define WAVE_BYTES 54016
#define CONST_BYTES 3712   // lnw[224] | lnb[128]@224 | b1[448]@352 | (bs+b2)[128]@800

// swizzled bf16 weight offsets in d_ws (elements)
#define OFF_WS0 0
#define OFF_WS1 16384
#define OFF_WS2 20480
#define OFF_W10 21504
#define OFF_W11 78848
#define OFF_W12 87040
#define OFF_W20 89088
#define OFF_W21 121856
#define OFF_W22 130048
// total = 132096 bf16 = 264192 bytes of workspace

__device__ __forceinline__ void wave_sync() {
  // intra-wave LDS producer->consumer ordering (no cross-wave sharing)
  asm volatile("s_wait_dscnt 0" ::: "memory");
}

__device__ __forceinline__ float fast_sigmoid(float s) {
  // v_exp_f32 + v_add + v_rcp_f32 (avoid IEEE divide expansion)
  return __builtin_amdgcn_rcpf(1.f + __expf(-s));
}

__device__ __forceinline__ v8f wmma_bf16(v16bf a, v16bf b, v8f c) {
  return __builtin_amdgcn_wmma_f32_16x16x32_bf16(false, a, false, b, (short)0, c,
                                                 false, false);
}

// B-fragment: pre-swizzled in global; each 32x16 (KxN) tile is a 1024B block,
// lane l owns 32 contiguous bytes (its 8 dwords, regs 0..7).
__device__ __forceinline__ v16bf loadB(const bf16* __restrict__ w, int blk, int lane) {
  const uint4* p = (const uint4*)(w + ((size_t)blk << 9) + (lane << 4));
  union { uint4 u[2]; v16bf v; } t;
  t.u[0] = p[0];
  t.u[1] = p[1];
  return t.v;
}

// A-fragment k index for reg r (16-bit A 16x32 layout; khalf8 = 8 for lanes 16..31)
__device__ __forceinline__ int kmap(int r, int khalf8) {
  return ((r < 4) ? 2 * r : 16 + 2 * (r - 4)) + khalf8;
}

__device__ __forceinline__ v16bf loadA_raw(const bf16* row, int base, int kstride,
                                           int kbase, int khalf8) {
  v16bf a;
#pragma unroll
  for (int r = 0; r < 8; ++r) {
    int k = kbase + kmap(r, khalf8);
    a[2 * r]     = row[base + k * kstride];
    a[2 * r + 1] = row[base + (k + 1) * kstride];
  }
  return a;
}

template <bool HASB>
__device__ __forceinline__ v16bf loadA_h(const bf16* row, int base, int kstride,
                                         int kbase, int khalf8, float mu, float rsq,
                                         const float* lnw, const float* lnb) {
  v16bf a;
#pragma unroll
  for (int r = 0; r < 8; ++r) {
    int k = kbase + kmap(r, khalf8);
    float x0 = (float)row[base + k * kstride];
    float x1 = (float)row[base + (k + 1) * kstride];
    float h0 = (x0 - mu) * rsq * lnw[k];
    float h1 = (x1 - mu) * rsq * lnw[k + 1];
    if constexpr (HASB) { h0 += lnb[k]; h1 += lnb[k + 1]; }
    a[2 * r]     = (bf16)h0;
    a[2 * r + 1] = (bf16)h1;
  }
  return a;
}

// ---- prep: convert W (KxN row-major f32) to bf16 fragment-ordered, scale folded ----
__global__ void swizzle_weights(const float* __restrict__ src, bf16* __restrict__ dst,
                                int K, int Nn, float scale, int total_dwords) {
  int od = blockIdx.x * blockDim.x + threadIdx.x;
  if (od >= total_dwords) return;
  int blk = od >> 8;        // 256 dwords per 32x16 tile
  int within = od & 255;
  int l = within >> 3;      // lane
  int r = within & 7;       // reg
  int kt_per = K >> 5;
  int ntile = blk / kt_per;
  int ktile = blk % kt_per;
  int n = ntile * 16 + (l & 15);
  int k = ktile * 32 + ((r < 4) ? 2 * r : 16 + 2 * (r - 4)) + ((l >> 4) & 1) * 8;
  dst[(size_t)od * 2]     = (bf16)(src[(size_t)k * Nn + n] * scale);
  dst[(size_t)od * 2 + 1] = (bf16)(src[(size_t)(k + 1) * Nn + n] * scale);
}

// ---- main fused kernel: one wave per 16-node tile ----
__global__ __launch_bounds__(128, 1) void eqffn_kernel(
    const float* __restrict__ xin, const float* __restrict__ attr,
    const bf16* __restrict__ W, const float* __restrict__ lnw_g,
    const float* __restrict__ lnb_g, const float* __restrict__ b1_g,
    const float* __restrict__ b2_g, const float* __restrict__ bs_g,
    float* __restrict__ out, int Nn, int ntiles) {
  extern __shared__ char smem[];
  float* consts = (float*)smem;
  for (int i = threadIdx.x; i < 224; i += blockDim.x) consts[i] = lnw_g[i];
  for (int i = threadIdx.x; i < 128; i += blockDim.x) consts[224 + i] = lnb_g[i];
  for (int i = threadIdx.x; i < 448; i += blockDim.x) consts[352 + i] = b1_g[i];
  for (int i = threadIdx.x; i < 128; i += blockDim.x) consts[800 + i] = bs_g[i] + b2_g[i];
  __syncthreads();

  const int lane = threadIdx.x & 31;
  const int wid = threadIdx.x >> 5;
  char* wb = smem + CONST_BYTES + wid * WAVE_BYTES;
  bf16* xs = (bf16*)wb;
  bf16* mid = (bf16*)(wb + 15616);
  bf16* gates = (bf16*)(wb + 15616 + 30976);
  float* stats = (float*)(wb + 15616 + 30976 + 6400);

  const int ln15 = lane & 15;
  const int hi = (lane >> 4) & 1;
  const int khalf8 = hi * 8;
  const bf16* arow = xs + ln15 * XSTR;
  const bf16* mrow = mid + ln15 * MSTR;
  const int wpb = blockDim.x >> 5;

  for (int tile = blockIdx.x * wpb + wid; tile < ntiles; tile += gridDim.x * wpb) {
    const int node0 = tile * 16;

    // ---- stage x tile: coalesced f32 global -> bf16 LDS ----
    for (int q = lane; q < 1920; q += 32) {
      int row = q / 120;
      int c4 = (q % 120) * 4;
      int gr = node0 + row;
      float4 v = make_float4(0.f, 0.f, 0.f, 0.f);
      if (gr < Nn) v = *(const float4*)(xin + (size_t)gr * 480 + c4);
      bf16* d = xs + row * XSTR + c4;
      d[0] = (bf16)v.x; d[1] = (bf16)v.y; d[2] = (bf16)v.z; d[3] = (bf16)v.w;
    }
    wave_sync();

    // ---- LN stats: all 32 lanes, lane = node(ln15) + 16*hi handles half the channels ----
    {
      const bf16* r = arow;
      float s0 = 0.f, ss0 = 0.f, ss1 = 0.f, ss2 = 0.f;
      int c0 = hi * 64;
#pragma unroll 16
      for (int c = 0; c < 64; ++c) {
        float x = (float)r[c0 + c];
        s0 += x;
        ss0 += x * x;
      }
      int c1 = 128 + hi * 96;
#pragma unroll 16
      for (int c = 0; c < 96; ++c) {
        float x = (float)r[c1 + c];
        ss1 += x * x;
      }
      int c2 = 320 + hi * 80;
#pragma unroll 16
      for (int c = 0; c < 80; ++c) {
        float x = (float)r[c2 + c];
        ss2 += x * x;
      }
      float* sp = stats + ln15 * SSTR + hi * 4;
      sp[0] = s0; sp[1] = ss0; sp[2] = ss1; sp[3] = ss2;
    }
    wave_sync();
    if (lane < 16) {
      float* sp = stats + lane * SSTR;
      float s0 = sp[0] + sp[4], ss0 = sp[1] + sp[5];
      float ss1 = sp[2] + sp[6], ss2 = sp[3] + sp[7];
      float mu = s0 * (1.f / 128.f);
      float var0 = ss0 * (1.f / 128.f) - mu * mu;
      sp[8] = mu;
      sp[9] = rsqrtf(var0 + 1e-5f);
      sp[10] = rsqrtf(ss1 * (1.f / 192.f) + 1e-5f);
      sp[11] = rsqrtf(ss2 * (1.f / 160.f) + 1e-5f);
      sp[12] = (node0 + lane < Nn) ? attr[node0 + lane] : 0.f;
    }
    wave_sync();

    const float mu = stats[ln15 * SSTR + 8];
    const float rq0 = stats[ln15 * SSTR + 9];
    const float rq1 = stats[ln15 * SSTR + 10];
    const float rq2 = stats[ln15 * SSTR + 11];

    // ---- g0 = h0(16x128) @ w1_0(128x448); silu -> scal, sigmoid -> gates ----
    {
      v16bf a[4];
#pragma unroll
      for (int kt = 0; kt < 4; ++kt)
        a[kt] = loadA_h<true>(arow, 0, 1, kt * 32, khalf8, mu, rq0, consts, consts + 224);
      // nt 0..15: cols 0..255 -> scal = silu(s)
      for (int nt = 0; nt < 16; ++nt) {
        v8f c = {};
#pragma unroll
        for (int kt = 0; kt < 4; ++kt)
          c = wmma_bf16(a[kt], loadB(W + OFF_W10, nt * 4 + kt, lane), c);
        int col = nt * 16 + ln15;
        float b1v = consts[352 + col];
#pragma unroll
        for (int r = 0; r < 8; ++r) {
          int node = r + hi * 8;
          float s = c[r] * stats[node * SSTR + 12] + b1v;
          mid[node * MSTR + col] = (bf16)(s * fast_sigmoid(s));
        }
      }
      // nt 16..27: cols 256..447 -> gates = sigmoid(s)
      for (int nt = 16; nt < 28; ++nt) {
        v8f c = {};
#pragma unroll
        for (int kt = 0; kt < 4; ++kt)
          c = wmma_bf16(a[kt], loadB(W + OFF_W10, nt * 4 + kt, lane), c);
        int col = nt * 16 + ln15;
        float b1v = consts[352 + col];
#pragma unroll
        for (int r = 0; r < 8; ++r) {
          int node = r + hi * 8;
          float s = c[r] * stats[node * SSTR + 12] + b1v;
          gates[node * GSTR + (col - 256)] = (bf16)fast_sigmoid(s);
        }
      }
    }
    wave_sync();

    // ---- g1 = h1m(16x64) @ w1_1(64x128); v1 = g1*gates[:,0:128] ----
    for (int m = 0; m < 3; ++m) {
      v16bf a0 = loadA_h<false>(arow, 128 + m, 3, 0, khalf8, 0.f, rq1, consts + 128, nullptr);
      v16bf a1 = loadA_h<false>(arow, 128 + m, 3, 32, khalf8, 0.f, rq1, consts + 128, nullptr);
      for (int nt = 0; nt < 8; ++nt) {
        v8f c = {};
        c = wmma_bf16(a0, loadB(W + OFF_W11, nt * 2 + 0, lane), c);
        c = wmma_bf16(a1, loadB(W + OFF_W11, nt * 2 + 1, lane), c);
        int col = nt * 16 + ln15;
#pragma unroll
        for (int r = 0; r < 8; ++r) {
          int node = r + hi * 8;
          float g = (float)gates[node * GSTR + col];
          mid[node * MSTR + 256 + m * 128 + col] = (bf16)(c[r] * stats[node * SSTR + 12] * g);
        }
      }
    }
    // ---- g2 = h2m(16x32) @ w1_2(32x64); v2 = g2*gates[:,128:192] ----
    for (int m = 0; m < 5; ++m) {
      v16bf a0 = loadA_h<false>(arow, 320 + m, 5, 0, khalf8, 0.f, rq2, consts + 192, nullptr);
      for (int nt = 0; nt < 4; ++nt) {
        v8f c = {};
        c = wmma_bf16(a0, loadB(W + OFF_W12, nt, lane), c);
        int col = nt * 16 + ln15;
#pragma unroll
        for (int r = 0; r < 8; ++r) {
          int node = r + hi * 8;
          float g = (float)gates[node * GSTR + 128 + col];
          mid[node * MSTR + 640 + m * 64 + col] = (bf16)(c[r] * stats[node * SSTR + 12] * g);
        }
      }
    }
    wave_sync();

    // ---- out seg0: x0@ws0 + scal@w2_0, +bs+b2, *attr ----
    {
      v16bf ax[4], am[8];
#pragma unroll
      for (int kt = 0; kt < 4; ++kt) ax[kt] = loadA_raw(arow, 0, 1, kt * 32, khalf8);
#pragma unroll
      for (int kt = 0; kt < 8; ++kt) am[kt] = loadA_raw(mrow, 0, 1, kt * 32, khalf8);
      for (int nt = 0; nt < 8; ++nt) {
        v8f c = {};
#pragma unroll
        for (int kt = 0; kt < 4; ++kt)
          c = wmma_bf16(ax[kt], loadB(W + OFF_WS0, nt * 4 + kt, lane), c);
#pragma unroll
        for (int kt = 0; kt < 8; ++kt)
          c = wmma_bf16(am[kt], loadB(W + OFF_W20, nt * 8 + kt, lane), c);
        int col = nt * 16 + ln15;
        float bv = consts[800 + col];
#pragma unroll
        for (int r = 0; r < 8; ++r) {
          int node = r + hi * 8;
          int gn = node0 + node;
          if (gn < Nn) out[(size_t)gn * 480 + col] = c[r] * stats[node * SSTR + 12] + bv;
        }
      }
    }
    // ---- out seg1: x1m@ws1 + v1m@w2_1, interleaved store ----
    for (int m = 0; m < 3; ++m) {
      v16bf ax[2], am[4];
#pragma unroll
      for (int kt = 0; kt < 2; ++kt) ax[kt] = loadA_raw(arow, 128 + m, 3, kt * 32, khalf8);
#pragma unroll
      for (int kt = 0; kt < 4; ++kt) am[kt] = loadA_raw(mrow, 256 + m * 128, 1, kt * 32, khalf8);
      for (int nt = 0; nt < 4; ++nt) {
        v8f c = {};
#pragma unroll
        for (int kt = 0; kt < 2; ++kt)
          c = wmma_bf16(ax[kt], loadB(W + OFF_WS1, nt * 2 + kt, lane), c);
#pragma unroll
        for (int kt = 0; kt < 4; ++kt)
          c = wmma_bf16(am[kt], loadB(W + OFF_W21, nt * 4 + kt, lane), c);
        int col = nt * 16 + ln15;
#pragma unroll
        for (int r = 0; r < 8; ++r) {
          int node = r + hi * 8;
          int gn = node0 + node;
          if (gn < Nn) out[(size_t)gn * 480 + 128 + col * 3 + m] = c[r] * stats[node * SSTR + 12];
        }
      }
    }
    // ---- out seg2: x2m@ws2 + v2m@w2_2 ----
    for (int m = 0; m < 5; ++m) {
      v16bf ax = loadA_raw(arow, 320 + m, 5, 0, khalf8);
      v16bf am0 = loadA_raw(mrow, 640 + m * 64, 1, 0, khalf8);
      v16bf am1 = loadA_raw(mrow, 640 + m * 64, 1, 32, khalf8);
      for (int nt = 0; nt < 2; ++nt) {
        v8f c = {};
        c = wmma_bf16(ax, loadB(W + OFF_WS2, nt, lane), c);
        c = wmma_bf16(am0, loadB(W + OFF_W22, nt * 2 + 0, lane), c);
        c = wmma_bf16(am1, loadB(W + OFF_W22, nt * 2 + 1, lane), c);
        int col = nt * 16 + ln15;
#pragma unroll
        for (int r = 0; r < 8; ++r) {
          int node = r + hi * 8;
          int gn = node0 + node;
          if (gn < Nn) out[(size_t)gn * 480 + 320 + col * 5 + m] = c[r] * stats[node * SSTR + 12];
        }
      }
    }
    wave_sync();  // xs/mid reads complete before next tile overwrites
  }
}

extern "C" void kernel_launch(void* const* d_in, const int* in_sizes, int n_in,
                              void* d_out, int out_size, void* d_ws, size_t ws_size,
                              hipStream_t stream) {
  const float* xin = (const float*)d_in[0];
  const float* attr = (const float*)d_in[1];
  // d_in[2] = batch (unused by reference forward)
  const float* lnw = (const float*)d_in[3];
  const float* lnb = (const float*)d_in[4];
  const float* w10 = (const float*)d_in[5];
  const float* w11 = (const float*)d_in[6];
  const float* w12 = (const float*)d_in[7];
  const float* b1 = (const float*)d_in[8];
  const float* w20 = (const float*)d_in[9];
  const float* w21 = (const float*)d_in[10];
  const float* w22 = (const float*)d_in[11];
  const float* b2 = (const float*)d_in[12];
  const float* ws0 = (const float*)d_in[13];
  const float* ws1 = (const float*)d_in[14];
  const float* ws2 = (const float*)d_in[15];
  const float* bs = (const float*)d_in[16];

  bf16* W = (bf16*)d_ws;  // needs 264192 bytes
  auto swz = [&](const float* src, int off, int K, int N) {
    int td = K * N / 2;
    swizzle_weights<<<(td + 255) / 256, 256, 0, stream>>>(src, W + off, K, N,
                                                          1.0f / sqrtf((float)K), td);
  };
  swz(ws0, OFF_WS0, 128, 128);
  swz(ws1, OFF_WS1, 64, 64);
  swz(ws2, OFF_WS2, 32, 32);
  swz(w10, OFF_W10, 128, 448);
  swz(w11, OFF_W11, 64, 128);
  swz(w12, OFF_W12, 32, 64);
  swz(w20, OFF_W20, 256, 128);
  swz(w21, OFF_W21, 128, 64);
  swz(w22, OFF_W22, 64, 32);

  int Nn = in_sizes[0] / 480;
  int ntiles = (Nn + 15) / 16;
  const int wpb = 4;  // waves per block
  int blocks = (ntiles + wpb - 1) / wpb;
  size_t shmem = CONST_BYTES + (size_t)wpb * WAVE_BYTES;  // ~219.8 KB < 320 KB
  eqffn_kernel<<<blocks, wpb * 32, shmem, stream>>>(xin, attr, W, lnw, lnb, b1, b2, bs,
                                                    (float*)d_out, Nn, ntiles);
}
